// ParallelMistralAttention_16020228014980
// MI455X (gfx1250) — matrix operand order repaired and checked by reading, not compiled, not run on hardware
//
#include <hip/hip_runtime.h>
#include <stdint.h>
#include <stddef.h>

#define S_LEN 2048
#define HIDD  4096
#define NH    32
#define NKV   8
#define HD    128
#define QK_SCALE 0.08838834764831845f   // 1/sqrt(128)

typedef unsigned short u16;
typedef __attribute__((ext_vector_type(16))) __bf16 v16bf;
typedef __attribute__((ext_vector_type(8)))  __bf16 v8bf;
typedef __attribute__((ext_vector_type(8)))  float  v8f;

struct v8bf_pair { v8bf lo, hi; };
__device__ __forceinline__ v16bf cat8(v8bf lo, v8bf hi) {
  v8bf_pair t{lo, hi};
  return __builtin_bit_cast(v16bf, t);   // pure register-pair concat, no element moves
}

__device__ __forceinline__ u16 f2bf(float f) {
  union { float f; uint32_t u; } v; v.f = f;
  uint32_t r = (v.u + 0x7FFFu + ((v.u >> 16) & 1u)) >> 16;
  return (u16)r;
}

// ---------------- fp32 -> bf16 elementwise ----------------
__global__ void k_cvt_bf16(const float* __restrict__ src, u16* __restrict__ dst, int n) {
  int i = blockIdx.x * blockDim.x + threadIdx.x;
  if (i < n) dst[i] = f2bf(src[i]);
}

// ---------------- fp32 [K][N] -> bf16 [N][K] (transpose) ----------------
__global__ void k_transpose_bf16(const float* __restrict__ src, u16* __restrict__ dst,
                                 int K, int N) {
  int i = blockIdx.x * blockDim.x + threadIdx.x;  // i = n*K + k  (k fastest -> coalesced store)
  if (i >= K * N) return;
  int k = i % K;
  int n = i / K;
  dst[i] = f2bf(src[(size_t)k * N + n]);
}

// ---- bf16 GEMM: C[M,N] = A[M,K] * Bt[N,K]^T, fp32 out, register-blocked ----
// One wave computes a (16*BM) x (16*BN) C macro-tile; 16x16x32 WMMA micro-tiles.
// Per k-chunk of 32: BM+BN fragment loads feed BM*BN WMMAs.
template <int BM, int BN>
__global__ __launch_bounds__(128, 1)
void k_gemm_bf16(const u16* __restrict__ A, const u16* __restrict__ Bt,
                 float* __restrict__ C, int M, int N, int K) {
  int wave = (int)((blockIdx.x * blockDim.x + threadIdx.x) >> 5);
  int lane = threadIdx.x & 31;
  int tilesN = N / (16 * BN);
  int tm = wave / tilesN;
  int tn = wave - tm * tilesN;
  if (tm >= M / (16 * BM)) return;
  int m0 = tm * 16 * BM, n0 = tn * 16 * BN;
  int lhalf = lane >> 4;        // 0: lanes 0-15, 1: lanes 16-31
  int l16   = lane & 15;

  const u16* abase = A  + (size_t)(m0 + l16) * K;   // A rows m0+l16 + 16*i
  const u16* bbase = Bt + (size_t)(n0 + l16) * K;   // B cols n0+l16 + 16*j

  v8f acc[BM][BN];
#pragma unroll
  for (int i = 0; i < BM; ++i)
#pragma unroll
    for (int j = 0; j < BN; ++j) acc[i][j] = (v8f){};

#pragma unroll 1
  for (int k0 = 0; k0 < K; k0 += 32) {
    v16bf a[BM], b[BN];
#pragma unroll
    for (int i = 0; i < BM; ++i) {
      const u16* p = abase + (size_t)i * 16 * K + k0;
      // A 16x32 bf16 operand: lanes<16 hold K 0-15, lanes>=16 hold K 16-31
      v8bf lo = *(const v8bf*)(p +     lhalf * 16);
      v8bf hi = *(const v8bf*)(p + 8 + lhalf * 16);
      a[i] = cat8(lo, hi);
      __builtin_prefetch(p + 128, 0, 0);            // stream next chunks toward WGP
    }
#pragma unroll
    for (int j = 0; j < BN; ++j) {
      // B 32x16 bf16 operand: lane<16 column n, rows K 0..15; lane>=16 rows K 16..31
      const u16* p = bbase + (size_t)j * 16 * K + k0 + lhalf * 16;
      b[j] = *(const v16bf*)p;
      __builtin_prefetch(p + 128, 0, 0);
    }
#pragma unroll
    for (int i = 0; i < BM; ++i)
#pragma unroll
      for (int j = 0; j < BN; ++j)
        acc[i][j] = __builtin_amdgcn_wmma_f32_16x16x32_bf16(false, a[i], false, b[j],
                                                            (short)0, acc[i][j], false, false);
  }
  // C/D layout: VGPR g -> row (16*i + g + 8*lhalf), col = 16*j + (lane&15)
#pragma unroll
  for (int i = 0; i < BM; ++i)
#pragma unroll
    for (int j = 0; j < BN; ++j) {
      float* crow = C + (size_t)(m0 + 16 * i + 8 * lhalf) * N + n0 + 16 * j + l16;
#pragma unroll
      for (int g = 0; g < 8; ++g) crow[(size_t)g * N] = acc[i][j][g];
    }
}

// ---------------- RoPE + pack per-head bf16: src [S, H*128] -> dst [H][S][128]
__global__ void k_rope_pack(const float* __restrict__ src, const float* __restrict__ cosT,
                            const float* __restrict__ sinT, u16* __restrict__ dst, int H) {
  int idx = blockIdx.x * blockDim.x + threadIdx.x;   // over S * H * 64
  if (idx >= S_LEN * H * 64) return;
  int d = idx & 63;
  int h = (idx >> 6) % H;
  int s = idx / (64 * H);
  const float* r = src + (size_t)s * (H * HD) + (size_t)h * HD;
  float x0 = r[d], x1 = r[d + 64];
  float c0 = cosT[s * HD + d],      s0 = sinT[s * HD + d];
  float c1 = cosT[s * HD + d + 64], s1 = sinT[s * HD + d + 64];
  u16* w = dst + ((size_t)h * S_LEN + s) * HD;
  w[d]      = f2bf(x0 * c0 - x1 * s0);
  w[d + 64] = f2bf(x1 * c1 + x0 * s1);
}

// ---------------- V pack transposed: src [S, NKV*128] -> dst [NKV][128][S] ----
__global__ void k_pack_vt(const float* __restrict__ src, u16* __restrict__ dst) {
  int idx = blockIdx.x * blockDim.x + threadIdx.x;   // over S * NKV * 128
  if (idx >= S_LEN * NKV * HD) return;
  int d = idx & (HD - 1);
  int h = (idx >> 7) & (NKV - 1);
  int s = idx / (HD * NKV);
  dst[((size_t)h * HD + d) * S_LEN + s] = f2bf(src[(size_t)s * (NKV * HD) + h * HD + d]);
}

// ---------------- flash attention (causal, GQA) ----------------
// one wave per (head, 16-query block); 4 waves / 128-thread block (no cross-wave sync)
__global__ __launch_bounds__(128, 1)
void k_attn(const u16* __restrict__ Qb,  // [NH][S][128] bf16 (rope'd)
            const u16* __restrict__ Kb,  // [NKV][S][128] bf16 (rope'd)
            const u16* __restrict__ Vt,  // [NKV][128][S] bf16
            u16* __restrict__ ctx) {     // [S][NH*128] bf16
  __shared__ __align__(64) u16 lds[4][16 * 32];     // per-wave P tile, row-major [16][32]
  int wavein = threadIdx.x >> 5;
  int lane   = threadIdx.x & 31;
  int wg     = blockIdx.x * 4 + wavein;             // 0 .. NH*(S/16)-1
  int head   = wg >> 7;                             // S/16 == 128 blocks per head
  int qb     = wg & 127;
  int hkv    = head >> 2;                           // GQA group of 4
  int q0     = qb << 4;
  int lhalf  = lane >> 4;
  int l16    = lane & 15;
  u16* myl   = lds[wavein];

  // Preload Q as WMMA A-operands for all 4 K-chunks of D=128
  const u16* qrow = Qb + ((size_t)head * S_LEN + q0 + l16) * HD;
  v16bf qa[4];
#pragma unroll
  for (int c = 0; c < 4; ++c) {
    v8bf lo = *(const v8bf*)(qrow + 32 * c +     lhalf * 16);
    v8bf hi = *(const v8bf*)(qrow + 32 * c + 8 + lhalf * 16);
    qa[c] = cat8(lo, hi);
  }

  float m_i[8], l_i[8], alpha[8];
  v8f o[8];
#pragma unroll
  for (int g = 0; g < 8; ++g) { m_i[g] = -1e30f; l_i[g] = 0.f; o[g] = (v8f){}; }

  const u16* kbase = Kb + (size_t)hkv * S_LEN * HD;
  const u16* vbase = Vt + (size_t)hkv * HD * S_LEN;

  int kend = q0 + 16;
  for (int key0 = 0; key0 < kend; key0 += 32) {
    // --- scores: two 16-key N tiles ---
    v8f sc0 = {}, sc1 = {};
    {
      const u16* krow0 = kbase + (size_t)(key0 + l16) * HD;
      const u16* krow1 = kbase + (size_t)(key0 + 16 + l16) * HD;
#pragma unroll
      for (int c = 0; c < 4; ++c) {
        v16bf b0 = *(const v16bf*)(krow0 + 32 * c + lhalf * 16);
        sc0 = __builtin_amdgcn_wmma_f32_16x16x32_bf16(false, qa[c], false, b0,
                                                      (short)0, sc0, false, false);
      }
#pragma unroll
      for (int c = 0; c < 4; ++c) {
        v16bf b1 = *(const v16bf*)(krow1 + 32 * c + lhalf * 16);
        sc1 = __builtin_amdgcn_wmma_f32_16x16x32_bf16(false, qa[c], false, b1,
                                                      (short)0, sc1, false, false);
      }
    }
    // --- online softmax (per row g, rows = q0 + g + 8*lhalf) ---
#pragma unroll
    for (int g = 0; g < 8; ++g) {
      int qrow_i = q0 + g + 8 * lhalf;
      int key_a  = key0 + l16;
      int key_b  = key0 + 16 + l16;
      float sa = sc0[g] * QK_SCALE; if (key_a > qrow_i) sa = -1e30f;
      float sb = sc1[g] * QK_SCALE; if (key_b > qrow_i) sb = -1e30f;
      float rm = fmaxf(sa, sb);
#pragma unroll
      for (int off = 1; off < 16; off <<= 1) rm = fmaxf(rm, __shfl_xor(rm, off, 32));
      float mn = fmaxf(m_i[g], rm);
      float a  = __expf(m_i[g] - mn);
      float pa = __expf(sa - mn);
      float pb = __expf(sb - mn);
      float rs = pa + pb;
#pragma unroll
      for (int off = 1; off < 16; off <<= 1) rs += __shfl_xor(rs, off, 32);
      l_i[g] = l_i[g] * a + rs;
      m_i[g] = mn;
      alpha[g] = a;
      // write P into LDS row-major [16 rows][32 keys] (bf16)
      myl[(g + 8 * lhalf) * 32 + l16]      = f2bf(pa);
      myl[(g + 8 * lhalf) * 32 + 16 + l16] = f2bf(pb);
    }
    // rescale O accumulators
#pragma unroll
    for (int dt = 0; dt < 8; ++dt)
#pragma unroll
      for (int g = 0; g < 8; ++g) o[dt][g] *= alpha[g];

    // wait for this wave's LDS stores, then reload P in A-operand layout
    asm volatile("s_wait_dscnt 0x0" ::: "memory");
    v16bf pA;
    {
      v8bf lo = *(const v8bf*)(myl + l16 * 32 +     lhalf * 16);
      v8bf hi = *(const v8bf*)(myl + l16 * 32 + 8 + lhalf * 16);
      pA = cat8(lo, hi);
    }
    // --- O += P * V : 8 d-tiles of 16 ---
#pragma unroll
    for (int dt = 0; dt < 8; ++dt) {
      const u16* vrow = vbase + (size_t)(dt * 16 + l16) * S_LEN + key0 + lhalf * 16;
      v16bf vb = *(const v16bf*)vrow;
      o[dt] = __builtin_amdgcn_wmma_f32_16x16x32_bf16(false, pA, false, vb,
                                                      (short)0, o[dt], false, false);
    }
  }

  // epilogue: normalize and scatter ctx (bf16) at [q0+g+8*lhalf][head*128 + dt*16 + l16]
  float inv[8];
#pragma unroll
  for (int g = 0; g < 8; ++g) inv[g] = 1.0f / l_i[g];
  u16* crow = ctx + (size_t)(q0 + 8 * lhalf) * (NH * HD) + (size_t)head * HD + l16;
#pragma unroll
  for (int dt = 0; dt < 8; ++dt)
#pragma unroll
    for (int g = 0; g < 8; ++g)
      crow[(size_t)g * (NH * HD) + dt * 16] = f2bf(o[dt][g] * inv[g]);
}

// =====================================================================
extern "C" void kernel_launch(void* const* d_in, const int* in_sizes, int n_in,
                              void* d_out, int out_size, void* d_ws, size_t ws_size,
                              hipStream_t stream) {
  const float* hidden = (const float*)d_in[0];
  const float* cosT   = (const float*)d_in[1];
  const float* sinT   = (const float*)d_in[2];
  // d_in[3] = causal_mask (unused; causal handled analytically)
  const float* Wq = (const float*)d_in[4];
  const float* Wk = (const float*)d_in[5];
  const float* Wv = (const float*)d_in[6];
  const float* Wo = (const float*)d_in[7];
  float* out = (float*)d_out;

  char* ws = (char*)d_ws;
  auto alloc = [&](size_t bytes) -> char* {
    char* p = ws; ws += (bytes + 255) & ~(size_t)255; return p;
  };
  u16*   hid_bf = (u16*)alloc((size_t)S_LEN * HIDD * 2);
  u16*   wqT    = (u16*)alloc((size_t)HIDD * NH  * HD * 2);
  u16*   wkT    = (u16*)alloc((size_t)HIDD * NKV * HD * 2);
  u16*   wvT    = (u16*)alloc((size_t)HIDD * NKV * HD * 2);
  u16*   woT    = (u16*)alloc((size_t)(NH * HD) * HIDD * 2);
  float* qf     = (float*)alloc((size_t)S_LEN * NH  * HD * 4);
  float* kf     = (float*)alloc((size_t)S_LEN * NKV * HD * 4);
  float* vf     = (float*)alloc((size_t)S_LEN * NKV * HD * 4);
  u16*   q_bf   = (u16*)alloc((size_t)NH  * S_LEN * HD * 2);
  u16*   k_bf   = (u16*)alloc((size_t)NKV * S_LEN * HD * 2);
  u16*   vt_bf  = (u16*)alloc((size_t)NKV * HD * S_LEN * 2);
  u16*   ctx_bf = (u16*)alloc((size_t)S_LEN * NH * HD * 2);

  const int T = 256;
  // 1) conversions / transposes
  k_cvt_bf16<<<(S_LEN * HIDD) / T, T, 0, stream>>>(hidden, hid_bf, S_LEN * HIDD);
  k_transpose_bf16<<<(HIDD * NH  * HD) / T, T, 0, stream>>>(Wq, wqT, HIDD, NH  * HD);
  k_transpose_bf16<<<(HIDD * NKV * HD) / T, T, 0, stream>>>(Wk, wkT, HIDD, NKV * HD);
  k_transpose_bf16<<<(HIDD * NKV * HD) / T, T, 0, stream>>>(Wv, wvT, HIDD, NKV * HD);
  k_transpose_bf16<<<((NH * HD) * HIDD) / T, T, 0, stream>>>(Wo, woT, NH * HD, HIDD);

  // 2) projection GEMMs (register-blocked macro-tiles, 4 waves per 128-thread block)
  {
    // Q: 64x64 per wave -> (2048/64)*(4096/64) = 2048 waves
    int wavesQ = (S_LEN / 64) * ((NH * HD) / 64);
    k_gemm_bf16<4, 4><<<wavesQ / 4, 128, 0, stream>>>(hid_bf, wqT, qf, S_LEN, NH * HD, HIDD);
    // K/V: 32x32 per wave -> (2048/32)*(1024/32) = 2048 waves (keeps grid occupancy up)
    int wavesKV = (S_LEN / 32) * ((NKV * HD) / 32);
    k_gemm_bf16<2, 2><<<wavesKV / 4, 128, 0, stream>>>(hid_bf, wkT, kf, S_LEN, NKV * HD, HIDD);
    k_gemm_bf16<2, 2><<<wavesKV / 4, 128, 0, stream>>>(hid_bf, wvT, vf, S_LEN, NKV * HD, HIDD);
  }

  // 3) RoPE + per-head packing
  k_rope_pack<<<(S_LEN * NH  * 64) / T, T, 0, stream>>>(qf, cosT, sinT, q_bf, NH);
  k_rope_pack<<<(S_LEN * NKV * 64) / T, T, 0, stream>>>(kf, cosT, sinT, k_bf, NKV);
  k_pack_vt  <<<(S_LEN * NKV * HD) / T, T, 0, stream>>>(vf, vt_bf);

  // 4) causal flash attention (GQA 32q/8kv heads, D=128)
  k_attn<<<(NH * (S_LEN / 16)) / 4, 128, 0, stream>>>(q_bf, k_bf, vt_bf, ctx_bf);

  // 5) output projection -> fp32 d_out (64x64 per wave)
  {
    int wavesO = (S_LEN / 64) * (HIDD / 64);
    k_gemm_bf16<4, 4><<<wavesO / 4, 128, 0, stream>>>(ctx_bf, woT, out, S_LEN, HIDD, NH * HD);
  }
  (void)in_sizes; (void)n_in; (void)out_size; (void)ws_size;
}
